// VectorQuantizerEMA_74509092651574
// MI455X (gfx1250) — matrix operand order repaired and checked by reading, not compiled, not run on hardware
//
#include <hip/hip_runtime.h>

// VQ nearest-codebook argmin for MI455X (gfx1250, wave32, WMMA).
//
//   inputs  : [16, 64, 64, 64] f32 NCHW  (C = D = 64 embedding dim)
//   codebook: [1024, 64] f32
//   out     : [65536] int32 argmin indices
//
// score(m,k) = ||e_k||^2 - 2 * x_m . e_k   (||x||^2 row-constant, dropped)
// Matmul done with v_wmma_f32_16x16x32_f16 using an f16 hi/lo split
// (xh*eh + xl*eh + xh*el) for ~fp32 accuracy at matrix-core rates.
//
// Round 2: fully unrolled tile loop + branchless cndmask min-updates so the
// argmin state lives in VGPRs (round 1 spilled to scratch with divergent
// v_cmpx/s_cbranch chains in the inner loop); LDS addresses advance by
// pointer increments instead of per-iteration multiplies.

typedef _Float16 v16h __attribute__((ext_vector_type(16)));
typedef _Float16 v8h  __attribute__((ext_vector_type(8)));
typedef float    v8f  __attribute__((ext_vector_type(8)));

#define D_DIM       64
#define K_CODES     1024
#define PASS_CODES  512            // LDS-staged codes per pass (2 passes)
#define LDS_STRIDE  72             // halfs per row: 144 B = 36 dwords (bank-friendly, 16B aligned)
#define HW          4096           // 64*64
#define CHW         (64 * 4096)
#define THREADS     256            // 8 waves
#define WAVES       8
#define TILES_PER_WAVE 2
#define NUM_WGS     256            // 256 WGs * 8 waves * 2 tiles * 16 rows = 65536 rows

__global__ __launch_bounds__(THREADS)
void VectorQuantizerEMA_74509092651574_kernel(const float* __restrict__ x,
                                              const float* __restrict__ cb,
                                              int* __restrict__ out)
{
    __shared__ alignas(16) _Float16 s_hi[PASS_CODES * LDS_STRIDE];
    __shared__ alignas(16) _Float16 s_lo[PASS_CODES * LDS_STRIDE];
    __shared__ float s_norm[PASS_CODES];

    const int tid   = threadIdx.x;
    const int wave  = tid >> 5;
    const int lane  = tid & 31;
    const int ln    = lane & 15;          // row (A) / code column (B/C) within tile
    const int koff  = (lane >> 4) << 3;   // A-operand K sub-block: 0 or 8
    const int bkoff = (lane & 16);        // B-operand K half: 0 or 16

    float minv[TILES_PER_WAVE][8];
    int   mini[TILES_PER_WAVE][8];
#pragma unroll
    for (int t = 0; t < TILES_PER_WAVE; ++t)
#pragma unroll
        for (int j = 0; j < 8; ++j) { minv[t][j] = 3.0e38f; mini[t][j] = 0; }

    for (int pass = 0; pass < K_CODES / PASS_CODES; ++pass) {
        __syncthreads();   // prior pass's readers done before restaging

        // ---- Stage codebook slice into LDS as f16 hi/lo (coalesced) ----
        const float* cbp = cb + pass * PASS_CODES * D_DIM;
#pragma unroll
        for (int i = 0; i < (PASS_CODES * D_DIM) / THREADS; ++i) {
            int e = tid + i * THREADS;
            float v = cbp[e];
            _Float16 h = (_Float16)v;
            _Float16 l = (_Float16)(v - (float)h);
            int c = e >> 6, d = e & 63;
            s_hi[c * LDS_STRIDE + d] = h;
            s_lo[c * LDS_STRIDE + d] = l;
        }
        __syncthreads();

        // ---- Per-code squared norms (from hi+lo, ~fp32 exact) ----
        for (int c = tid; c < PASS_CODES; c += THREADS) {
            float s = 0.0f;
            for (int d = 0; d < D_DIM; ++d) {
                float v = (float)s_hi[c * LDS_STRIDE + d] + (float)s_lo[c * LDS_STRIDE + d];
                s = fmaf(v, v, s);
            }
            s_norm[c] = s;
        }
        __syncthreads();

        // ---- Compute: each wave owns TILES_PER_WAVE row-tiles of 16 pixels ----
#pragma unroll
        for (int t = 0; t < TILES_PER_WAVE; ++t) {
            const int T   = (blockIdx.x * WAVES + wave) * TILES_PER_WAVE + t; // 0..4095
            const int b   = T >> 8;                 // image: T*16 / 4096
            const int hwb = (T << 4) & (HW - 1);    // pixel base within image
            const float* px = x + b * CHW + hwb + ln;   // this lane's row m = ln

            // A operand: ISA 16-bit A 16x32 layout. lane half selects koff (0/8);
            // vector elems 0-7 = K=kb+koff.., elems 8-15 = K=kb+16+koff..
            v16h ah[2], al[2];
#pragma unroll
            for (int kc = 0; kc < 2; ++kc) {
                const int kb = kc * 32;
#pragma unroll
                for (int i = 0; i < 8; ++i) {
                    float v0 = px[(kb + koff + i) * HW];
                    float v1 = px[(kb + 16 + koff + i) * HW];
                    _Float16 h0 = (_Float16)v0;
                    _Float16 h1 = (_Float16)v1;
                    ah[kc][i]     = h0;  al[kc][i]     = (_Float16)(v0 - (float)h0);
                    ah[kc][8 + i] = h1;  al[kc][8 + i] = (_Float16)(v1 - (float)h1);
                }
            }

            // Incremented pointers: per ct step, code += 16 -> +16*LDS_STRIDE halfs.
            const _Float16* rh = &s_hi[ln * LDS_STRIDE + koff];
            const _Float16* rl = &s_lo[ln * LDS_STRIDE + koff];
            const float*    rn = &s_norm[ln];
            int gcode = pass * PASS_CODES + ln;     // this lane's candidate code

            for (int ct = 0; ct < PASS_CODES / 16; ++ct) {
                v8f acc = {};
#pragma unroll
                for (int kc = 0; kc < 2; ++kc) {
                    const int kb = kc * 32;
                    // B operand: lane = column, K order matches A (koff, 16+koff)
                    v16h bh, bl;
                    ((v8h*)&bh)[0] = *(const v8h*)(rh + kb);
                    ((v8h*)&bh)[1] = *(const v8h*)(rh + kb + 16);
                    ((v8h*)&bl)[0] = *(const v8h*)(rl + kb);
                    ((v8h*)&bl)[1] = *(const v8h*)(rl + kb + 16);

                    acc = __builtin_amdgcn_wmma_f32_16x16x32_f16(
                              false, ah[kc], false, bh, (short)0, acc, false, false);
                    acc = __builtin_amdgcn_wmma_f32_16x16x32_f16(
                              false, al[kc], false, bh, (short)0, acc, false, false);
                    acc = __builtin_amdgcn_wmma_f32_16x16x32_f16(
                              false, ah[kc], false, bl, (short)0, acc, false, false);
                }

                const float nrm = *rn;
#pragma unroll
                for (int j = 0; j < 8; ++j) {       // C: VGPR j = row (j + 8*lanehalf)
                    float dist = fmaf(-2.0f, acc[j], nrm);
                    bool better = dist < minv[t][j];        // strict '<' keeps first min
                    minv[t][j] = better ? dist  : minv[t][j];   // v_cndmask
                    mini[t][j] = better ? gcode : mini[t][j];   // v_cndmask
                }

                rh += 16 * LDS_STRIDE;
                rl += 16 * LDS_STRIDE;
                rn += 16;
                gcode += 16;
            }
        }
    }

    // ---- Cross-lane argmin (xor 1,2,4,8 stays inside each 16-lane half) ----
#pragma unroll
    for (int t = 0; t < TILES_PER_WAVE; ++t) {
        const int T = (blockIdx.x * WAVES + wave) * TILES_PER_WAVE + t;
#pragma unroll
        for (int j = 0; j < 8; ++j) {
            float v  = minv[t][j];
            int   ix = mini[t][j];
#pragma unroll
            for (int s = 1; s < 16; s <<= 1) {
                float ov = __shfl_xor(v, s, 32);
                int   oi = __shfl_xor(ix, s, 32);
                bool take = (ov < v) || (ov == v && oi < ix);  // first-min tie-break
                v  = take ? ov : v;
                ix = take ? oi : ix;
            }
            if (ln == 0) {   // lanes 0 (rows 0-7) and 16 (rows 8-15)
                out[T * 16 + ((lane >> 4) << 3) + j] = ix;
            }
        }
    }
}

extern "C" void kernel_launch(void* const* d_in, const int* in_sizes, int n_in,
                              void* d_out, int out_size, void* d_ws, size_t ws_size,
                              hipStream_t stream) {
    const float* x  = (const float*)d_in[0];   // [16,64,64,64] f32
    const float* cb = (const float*)d_in[1];   // [1024,64] f32
    int* out = (int*)d_out;                    // [65536] int32 indices
    VectorQuantizerEMA_74509092651574_kernel<<<dim3(NUM_WGS), dim3(THREADS), 0, stream>>>(
        x, cb, out);
}
